// ADGN_6253472383693
// MI455X (gfx1250) — matrix operand order repaired and checked by reading, not compiled, not run on hardware
//
#include <hip/hip_runtime.h>

typedef __attribute__((ext_vector_type(16))) _Float16 v16h;
typedef __attribute__((ext_vector_type(8)))  float    v8f;

#define HID 128
#define GAMMA 0.1f
#define EPS 0.1f

// ---------------- weight prep: comb_w = [lin_w ; W - W^T - gamma*I] as f16, emb_w as f16 ----
__global__ void adgn_prep_weights(const float* __restrict__ W,
                                  const float* __restrict__ lin_w,
                                  const float* __restrict__ emb_w,
                                  _Float16* __restrict__ comb_w,   // [256][128]
                                  _Float16* __restrict__ emb_wh) { // [128][64]
  int idx = blockIdx.x * blockDim.x + threadIdx.x;
  const int NC = 256 * 128;
  if (idx < NC) {
    int n = idx >> 7, k = idx & 127;
    float v;
    if (n < 128) {
      v = lin_w[n * 128 + k];
    } else {
      int r = n - 128;
      v = W[r * 128 + k] - W[k * 128 + r] - ((r == k) ? GAMMA : 0.f);
    }
    comb_w[idx] = (_Float16)v;
  } else if (idx < NC + 128 * 64) {
    int i = idx - NC;
    emb_wh[i] = (_Float16)emb_w[i];
  }
}

// ---------------- WMMA GEMM: C[M, NT*16] = A[M,K] * B^T (+bias), B row-major [NT*16][K] f16 --
// One wave32 owns a 16-row tile. A fragments in contiguous k order (k = 16*h + s, same as B), reused for all
// NT column tiles. Two independent accumulator chains are interleaved so the second chain's
// v_wmma fills the first chain's RAW-hazard slots (replaces the compiler's v_nops).
template <int K, int NT, bool HAS_BIAS>
__global__ void adgn_gemm_wmma(const float* __restrict__ A,
                               const _Float16* __restrict__ B,
                               const float* __restrict__ bias,
                               float* __restrict__ C, int M) {
  constexpr int N  = NT * 16;
  constexpr int KS = K / 32;
  const int wave = threadIdx.x >> 5;
  const int lane = threadIdx.x & 31;
  const int rowbase = (blockIdx.x * (blockDim.x >> 5) + wave) * 16;
  if (rowbase >= M) return;                 // uniform per wave; M % 16 == 0
  const int  l15 = lane & 15;
  const bool hi  = lane >= 16;

  // Load + convert this wave's 16 rows of A into K/32 fragments.
  const float* arow = A + (size_t)(rowbase + l15) * K;
  v16h afrag[KS];
#pragma unroll
  for (int ks = 0; ks < KS; ++ks) {
    const int k0 = ks * 32 + (hi ? 16 : 0);
    v16h f;
#pragma unroll
    for (int i = 0; i < 8; ++i) f[i]     = (_Float16)arow[k0 + i];
#pragma unroll
    for (int i = 0; i < 8; ++i) f[8 + i] = (_Float16)arow[k0 + 8 + i];
    afrag[ks] = f;
  }

  const int r0   = rowbase + (hi ? 8 : 0);
  const int boff = hi ? 16 : 0;
#pragma unroll
  for (int np = 0; np < NT; np += 2) {      // two n-tiles at a time (NT is even)
    const int col0 = np * 16 + l15;
    const int col1 = col0 + 16;
    v8f acc0, acc1;
    const float bv0 = HAS_BIAS ? bias[col0] : 0.f;
    const float bv1 = HAS_BIAS ? bias[col1] : 0.f;
#pragma unroll
    for (int r = 0; r < 8; ++r) { acc0[r] = bv0; acc1[r] = bv1; }
#pragma unroll
    for (int ks = 0; ks < KS; ++ks) {
      const v16h bf0 = *(const v16h*)(B + (size_t)col0 * K + ks * 32 + boff);
      const v16h bf1 = *(const v16h*)(B + (size_t)col1 * K + ks * 32 + boff);
      acc0 = __builtin_amdgcn_wmma_f32_16x16x32_f16(false, afrag[ks], false, bf0,
                                                    (short)0, acc0, false, false);
      acc1 = __builtin_amdgcn_wmma_f32_16x16x32_f16(false, afrag[ks], false, bf1,
                                                    (short)0, acc1, false, false);
    }
#pragma unroll
    for (int r = 0; r < 8; ++r) {
      C[(size_t)(r0 + r) * N + col0] = acc0[r];
      C[(size_t)(r0 + r) * N + col1] = acc1[r];
    }
  }
}

// ---------------- zero ----------------
__global__ void adgn_zero(float* __restrict__ p, int n) {
  int i = blockIdx.x * blockDim.x + threadIdx.x;
  if (i < n) p[i] = 0.f;
}

// ---------------- edge scatter-add: one wave per edge, float4 per lane -------------------
__global__ void adgn_scatter(const float* __restrict__ combo,   // [M][256], cols 0..127 = hl
                             const long long* __restrict__ src,
                             const long long* __restrict__ dst,
                             float* __restrict__ neigh, int E) {
  int t = blockIdx.x * blockDim.x + threadIdx.x;
  int e = t >> 5;
  int lane = t & 31;
  if (e >= E) return;
  const long long s = src[e];
  const long long d = dst[e];
  const float4 v = *(const float4*)(combo + (size_t)s * 256 + lane * 4);
  float* np = neigh + (size_t)d * 128 + lane * 4;
  atomicAdd(np + 0, v.x);
  atomicAdd(np + 1, v.y);
  atomicAdd(np + 2, v.z);
  atomicAdd(np + 3, v.w);
}

// ---------------- h += eps * tanh(h@aW^T + neigh + asym_b) -------------------------------
__global__ void adgn_update(float* __restrict__ h, const float* __restrict__ combo,
                            const float* __restrict__ neigh,
                            const float* __restrict__ asym_b, int M) {
  int i = blockIdx.x * blockDim.x + threadIdx.x;
  if (i >= M * HID) return;
  int row = i >> 7, c = i & 127;
  float conv = combo[(size_t)row * 256 + 128 + c] + neigh[i] + asym_b[c];
  h[i] += EPS * tanhf(conv);
}

// ---------------- pooling: batch is sorted, 1 block per graph, 1 thread per channel ------
__device__ __forceinline__ int lb_ll(const long long* a, int n, long long v) {
  int lo = 0, hi = n;
  while (lo < hi) { int mid = (lo + hi) >> 1; if (a[mid] < v) lo = mid + 1; else hi = mid; }
  return lo;
}

__global__ void adgn_pool(const float* __restrict__ h, const long long* __restrict__ batch,
                          float* __restrict__ g3, int Nn) {
  int g = blockIdx.x;
  int c = threadIdx.x;           // 128 threads
  int start = lb_ll(batch, Nn, (long long)g);
  int end   = lb_ll(batch, Nn, (long long)g + 1);
  float add = 0.f, mx = -3.402823466e+38f;
  for (int n = start; n < end; ++n) {
    float v = h[(size_t)n * HID + c];
    add += v;
    mx = fmaxf(mx, v);
  }
  float cnt  = (float)(end - start);
  float mean = add / fmaxf(cnt, 1.f);
  g3[g * 384 + c]       = add;
  g3[g * 384 + 128 + c] = mx;
  g3[g * 384 + 256 + c] = mean;
}

// ---------------- tiny readout MLP ----------------
__global__ void adgn_mlp1(const float* __restrict__ g3, const float* __restrict__ w,
                          const float* __restrict__ b, float* __restrict__ y1) {
  __shared__ float row[384];
  int g = blockIdx.x, j = threadIdx.x;  // 192 threads
  for (int k = threadIdx.x; k < 384; k += blockDim.x) row[k] = g3[g * 384 + k];
  __syncthreads();
  float acc = b[j];
  for (int k = 0; k < 384; ++k) acc += row[k] * w[j * 384 + k];
  y1[g * 192 + j] = acc > 0.f ? acc : 0.01f * acc;
}

__global__ void adgn_mlp2(const float* __restrict__ y1, const float* __restrict__ w,
                          const float* __restrict__ b, float* __restrict__ out) {
  int g = blockIdx.x, o = threadIdx.x;  // 16 threads
  float acc = b[o];
  for (int k = 0; k < 192; ++k) acc += y1[g * 192 + k] * w[o * 192 + k];
  out[g * 16 + o] = acc > 0.f ? acc : 0.01f * acc;
}

// ---------------- launch ----------------
extern "C" void kernel_launch(void* const* d_in, const int* in_sizes, int n_in,
                              void* d_out, int out_size, void* d_ws, size_t ws_size,
                              hipStream_t stream) {
  const float*     x      = (const float*)d_in[0];
  const long long* edge   = (const long long*)d_in[1];
  const long long* batch  = (const long long*)d_in[2];
  const float*     emb_w  = (const float*)d_in[3];
  const float*     emb_b  = (const float*)d_in[4];
  const float*     W      = (const float*)d_in[5];
  const float*     asym_b = (const float*)d_in[6];
  const float*     lin_w  = (const float*)d_in[7];
  const float*     r1w    = (const float*)d_in[8];
  const float*     r1b    = (const float*)d_in[9];
  const float*     r2w    = (const float*)d_in[10];
  const float*     r2b    = (const float*)d_in[11];
  float* out = (float*)d_out;

  const int Nn = in_sizes[0] / 64;   // 100000 nodes
  const int E  = in_sizes[1] / 2;    // 1600000 edges

  char* ws = (char*)d_ws;
  float* h     = (float*)ws; ws += (size_t)Nn * 128 * 4;
  float* combo = (float*)ws; ws += (size_t)Nn * 256 * 4;   // cols 0..127 hl, 128..255 h@aW^T
  float* neigh = (float*)ws; ws += (size_t)Nn * 128 * 4;
  float* g3    = (float*)ws; ws += 128 * 384 * 4;
  float* y1    = (float*)ws; ws += 128 * 192 * 4;
  _Float16* comb_w = (_Float16*)ws; ws += 256 * 128 * 2;
  _Float16* emb_wh = (_Float16*)ws; ws += 128 * 64 * 2;

  // 0) weights
  {
    int tot = 256 * 128 + 128 * 64;
    adgn_prep_weights<<<(tot + 255) / 256, 256, 0, stream>>>(W, lin_w, emb_w, comb_w, emb_wh);
  }

  // 1) embed: h = x @ emb_w^T + emb_b      (M x 64) * (64 x 128)
  {
    int blocks = (Nn + 127) / 128;  // 8 waves/block * 16 rows
    adgn_gemm_wmma<64, 8, true><<<blocks, 256, 0, stream>>>(x, emb_wh, emb_b, h, Nn);
  }

  // 2) 4 message-passing iterations
  for (int it = 0; it < 4; ++it) {
    int blocks = (Nn + 127) / 128;
    adgn_gemm_wmma<128, 16, false><<<blocks, 256, 0, stream>>>(h, comb_w, nullptr, combo, Nn);
    adgn_zero<<<((Nn * 128) + 255) / 256, 256, 0, stream>>>(neigh, Nn * 128);
    adgn_scatter<<<(E + 7) / 8, 256, 0, stream>>>(combo, edge, edge + E, neigh, E);
    adgn_update<<<((Nn * 128) + 255) / 256, 256, 0, stream>>>(h, combo, neigh, asym_b, Nn);
  }

  // 3) pooling + readout
  adgn_pool<<<128, 128, 0, stream>>>(h, batch, g3, Nn);
  adgn_mlp1<<<128, 192, 0, stream>>>(g3, r1w, r1b, y1);
  adgn_mlp2<<<128, 16, 0, stream>>>(y1, r2w, r2b, out);
}
